// MultiHeadAttention_27968827031607
// MI455X (gfx1250) — matrix operand order repaired and checked by reading, not compiled, not run on hardware
//
#include <hip/hip_runtime.h>

// ---------------------------------------------------------------------------
// MHA forward for MI455X (gfx1250, wave32, WMMA 16x16x32 f16 -> f32 acc)
// B=2, S=2048, D=1024, H=16, hd=64
// ---------------------------------------------------------------------------

typedef __attribute__((ext_vector_type(16))) _Float16 v16h;
typedef __attribute__((ext_vector_type(8)))  float    v8f;
typedef int v4i_t __attribute__((vector_size(16)));

#define WMMA_F16(A, B, C) \
    __builtin_amdgcn_wmma_f32_16x16x32_f16(false, (A), false, (B), (short)0, (C), false, false)

// ---- gfx1250 async global->LDS staging (guarded; falls back to plain copy) --
#if defined(__has_builtin)
#if __has_builtin(__builtin_amdgcn_global_load_async_to_lds_b128) && \
    __has_builtin(__builtin_amdgcn_s_wait_asynccnt)
#define HAVE_ASYNC_LDS 1
#endif
#endif

__device__ __forceinline__ void async_copy_b128(void* dst, const void* src) {
#if defined(HAVE_ASYNC_LDS)
    // Signature (probe-derived): (global int4* src, local int4* dst, Ii, Ii)
    __builtin_amdgcn_global_load_async_to_lds_b128(
        (__attribute__((address_space(1))) v4i_t*)(v4i_t*)const_cast<void*>(src),
        (__attribute__((address_space(3))) v4i_t*)(v4i_t*)dst,
        0, 0);
#else
    *(float4*)dst = *(const float4*)src;
#endif
}

__device__ __forceinline__ void async_join() {
#if defined(HAVE_ASYNC_LDS)
    __builtin_amdgcn_s_wait_asynccnt(0);
#endif
}

static constexpr int BATCH = 2;
static constexpr int SEQ   = 2048;
static constexpr int DMODEL= 1024;
static constexpr int HEADS = 16;
static constexpr int HDIM  = 64;

// ---------------------------------------------------------------------------
// Tiled GEMM:  Y[M,N] = cast<OT>( X[M,K] @ W[K,N] + bias[N] )
// Block: 128 threads (4 waves). 64x64 output tile, 64-wide K slice.
// Each wave: 16 rows x 64 cols, 8 WMMAs per K step. A staged row-major f16;
// B staged *transposed* f16 so all fragment loads are contiguous ds_load_b128
// per the CDNA5 ISA 16-bit A/B VGPR layouts.
// ---------------------------------------------------------------------------
template <typename AT, typename OT>
__global__ __launch_bounds__(128) void gemm_wmma(const AT* __restrict__ X,
                                                 const float* __restrict__ W,
                                                 const float* __restrict__ bias,
                                                 OT* __restrict__ Y,
                                                 int M, int N, int K) {
    __shared__ _Float16 As[64][64];    // [row][k]  within current k-slice
    __shared__ _Float16 BsT[64][72];   // [col][k]  (transposed, padded)

    const int t    = threadIdx.x;
    const int w    = t >> 5;           // wave id 0..3
    const int lane = t & 31;
    const int r    = lane & 15;
    const int hi   = lane >> 4;

    const int tileM = blockIdx.y * 64;
    const int tileN = blockIdx.x * 64;

    v8f acc[4] = {v8f{}, v8f{}, v8f{}, v8f{}};

    for (int k0 = 0; k0 < K; k0 += 64) {
        __syncthreads();
        // ---- stage A tile: 64 rows x 64 k (4096 elems, 32/thread) ----
        #pragma unroll
        for (int i = 0; i < 8; ++i) {
            int fi  = t + 128 * i;       // 0..1023 float4-chunks
            int row = fi >> 4;
            int c4  = (fi & 15) * 4;
            #pragma unroll
            for (int j = 0; j < 4; ++j)
                As[row][c4 + j] = (_Float16)X[(size_t)(tileM + row) * K + k0 + c4 + j];
        }
        // ---- stage B tile transposed: 64 k x 64 n ----
        #pragma unroll
        for (int i = 0; i < 8; ++i) {
            int fi = t + 128 * i;        // 0..1023
            int kr = fi >> 4;            // 0..63
            int c4 = (fi & 15) * 4;      // 0..60
            #pragma unroll
            for (int j = 0; j < 4; ++j)
                BsT[c4 + j][kr] = (_Float16)W[(size_t)(k0 + kr) * N + tileN + c4 + j];
        }
        // ---- prefetch next k-slice into cache (global_prefetch_b8) ----
        if (k0 + 64 < K) {
            __builtin_prefetch(&X[(size_t)(tileM + (t >> 1)) * K + k0 + 64 + (t & 1) * 32], 0, 0);
            __builtin_prefetch(&W[(size_t)(k0 + 64 + (t >> 1)) * N + tileN + (t & 1) * 32], 0, 0);
        }
        __syncthreads();

        // ---- A fragments (16x32 each, ISA layout: k=(idx>=8?16:0)+8*hi+(idx&7)) ----
        v16h a0, a1;
        const _Float16* ap = &As[w * 16 + r][0];
        #pragma unroll
        for (int i = 0; i < 8; ++i) {
            a0[i]     = ap[8 * hi + i];
            a0[i + 8] = ap[16 + 8 * hi + i];
            a1[i]     = ap[32 + 8 * hi + i];
            a1[i + 8] = ap[48 + 8 * hi + i];
        }
        // ---- 4 x (2 B fragments + 2 WMMAs) ----
        #pragma unroll
        for (int n = 0; n < 4; ++n) {
            v16h b0, b1;
            const _Float16* bp = &BsT[n * 16 + r][0];
            #pragma unroll
            for (int i = 0; i < 16; ++i) {
                b0[i] = bp[(i < 8 ? 8 * hi + i : 16 + 8 * hi + (i - 8))];
                b1[i] = bp[32 + (i < 8 ? 8 * hi + i : 16 + 8 * hi + (i - 8))];
            }
            acc[n] = WMMA_F16(a0, b0, acc[n]);
            acc[n] = WMMA_F16(a1, b1, acc[n]);
        }
    }

    // ---- epilogue: C/D layout lane holds col = n*16+r, rows m = v + 8*hi ----
    #pragma unroll
    for (int n = 0; n < 4; ++n) {
        int col = tileN + n * 16 + r;
        float bv = bias[col];
        #pragma unroll
        for (int v = 0; v < 8; ++v) {
            int row = tileM + w * 16 + v + 8 * hi;
            Y[(size_t)row * N + col] = (OT)(acc[n][v] + bv);
        }
    }
}

// ---------------------------------------------------------------------------
// Flash attention per (b,h): 64 queries/block, 4 waves x 16 queries.
// Q/K chunks staged via async global->LDS DMA (ASYNCcnt) when available.
// V stored transposed so the PV B-fragments are contiguous. Online softmax in
// fp32 with 16-lane butterflies (wave halves never mix, matching C/D layout).
// ---------------------------------------------------------------------------
__global__ __launch_bounds__(128) void attn_wmma(const _Float16* __restrict__ Q,
                                                 const _Float16* __restrict__ Km,
                                                 const _Float16* __restrict__ Vm,
                                                 const unsigned char* __restrict__ mask,
                                                 _Float16* __restrict__ Ctx) {
    __shared__ _Float16 Qlds[64][64];
    __shared__ _Float16 Klds[64][64];
    __shared__ _Float16 VldsT[64][72];      // [hd][key], padded
    __shared__ _Float16 Plds[4][16][64];    // per-wave P slice (C-layout -> A-layout)
    __shared__ float    mbias[64];

    const int t    = threadIdx.x;
    const int w    = t >> 5;
    const int lane = t & 31;
    const int r    = lane & 15;
    const int hi   = lane >> 4;

    const int bh    = blockIdx.y;
    const int b     = bh / HEADS;
    const int h     = bh % HEADS;
    const int qBase = blockIdx.x * 64;
    const int hcol  = h * HDIM;

    // ---- load Q tile (64 q x 64 hd) via async DMA, 4 x b128 per thread ----
    #pragma unroll
    for (int i = 0; i < 4; ++i) {
        int fi  = t + 128 * i;           // 0..511 8-half chunks
        int row = fi >> 3;
        int c8  = (fi & 7) * 8;
        async_copy_b128(&Qlds[row][c8],
                        &Q[(size_t)(b * SEQ + qBase + row) * DMODEL + hcol + c8]);
    }
    async_join();
    __syncthreads();

    // ---- Q fragments: A-layout 16x32, two slices covering hd 0..63 ----
    v16h aq0, aq1;
    {
        const _Float16* qp = &Qlds[w * 16 + r][0];
        #pragma unroll
        for (int i = 0; i < 8; ++i) {
            aq0[i]     = qp[8 * hi + i];
            aq0[i + 8] = qp[16 + 8 * hi + i];
            aq1[i]     = qp[32 + 8 * hi + i];
            aq1[i + 8] = qp[48 + 8 * hi + i];
        }
    }

    float mrow[8], lrow[8];
    #pragma unroll
    for (int v = 0; v < 8; ++v) { mrow[v] = -1e30f; lrow[v] = 0.0f; }
    v8f accO[4] = {v8f{}, v8f{}, v8f{}, v8f{}};

    for (int kc = 0; kc < SEQ; kc += 64) {
        __syncthreads();
        // ---- stage K chunk via async DMA; V chunk via register transpose ----
        #pragma unroll
        for (int i = 0; i < 4; ++i) {
            int fi  = t + 128 * i;
            int row = fi >> 3;
            int c8  = (fi & 7) * 8;
            async_copy_b128(&Klds[row][c8],
                            &Km[(size_t)(b * SEQ + kc + row) * DMODEL + hcol + c8]);
            const _Float16* vs = &Vm[(size_t)(b * SEQ + kc + row) * DMODEL + hcol + c8];
            #pragma unroll
            for (int j = 0; j < 8; ++j)
                VldsT[c8 + j][row] = vs[j];
        }
        if (t < 64) mbias[t] = mask[(size_t)b * SEQ + kc + t] ? 0.0f : -1e30f;
        async_join();
        __syncthreads();

        // ---- scores: S = (Q @ K^T) * 1/sqrt(hd) + mask_bias ----
        v8f sc[4];
        #pragma unroll
        for (int kn = 0; kn < 4; ++kn) {
            v16h bk0, bk1;
            const _Float16* kp = &Klds[kn * 16 + r][0];
            #pragma unroll
            for (int i = 0; i < 16; ++i) {
                bk0[i] = kp[(i < 8 ? 8 * hi + i : 16 + 8 * hi + (i - 8))];
                bk1[i] = kp[32 + (i < 8 ? 8 * hi + i : 16 + 8 * hi + (i - 8))];
            }
            v8f z = v8f{};
            z = WMMA_F16(aq0, bk0, z);
            z = WMMA_F16(aq1, bk1, z);
            sc[kn] = z;
        }
        #pragma unroll
        for (int kn = 0; kn < 4; ++kn) {
            float mb = mbias[kn * 16 + r];
            #pragma unroll
            for (int v = 0; v < 8; ++v) sc[kn][v] = sc[kn][v] * 0.125f + mb;
        }

        // ---- online softmax (rows m = v + 8*hi; reductions stay in halves) ----
        #pragma unroll
        for (int v = 0; v < 8; ++v) {
            float lm = fmaxf(fmaxf(sc[0][v], sc[1][v]), fmaxf(sc[2][v], sc[3][v]));
            #pragma unroll
            for (int off = 1; off < 16; off <<= 1) lm = fmaxf(lm, __shfl_xor(lm, off, 32));
            float mnew = fmaxf(mrow[v], lm);
            float corr = __expf(mrow[v] - mnew);
            float ps = 0.0f;
            #pragma unroll
            for (int kn = 0; kn < 4; ++kn) {
                float p = __expf(sc[kn][v] - mnew);
                sc[kn][v] = p;
                ps += p;
            }
            #pragma unroll
            for (int off = 1; off < 16; off <<= 1) ps += __shfl_xor(ps, off, 32);
            lrow[v] = lrow[v] * corr + ps;
            mrow[v] = mnew;
            #pragma unroll
            for (int n = 0; n < 4; ++n) accO[n][v] *= corr;
        }

        // ---- C-layout P -> per-wave LDS (same-wave DS ops are in-order) ----
        #pragma unroll
        for (int v = 0; v < 8; ++v) {
            int qm = v + 8 * hi;
            #pragma unroll
            for (int kn = 0; kn < 4; ++kn)
                Plds[w][qm][kn * 16 + r] = (_Float16)sc[kn][v];
        }

        // ---- P fragments (A-layout) + PV WMMAs ----
        v16h ap0, ap1;
        {
            const _Float16* pp = &Plds[w][r][0];
            #pragma unroll
            for (int i = 0; i < 8; ++i) {
                ap0[i]     = pp[8 * hi + i];
                ap0[i + 8] = pp[16 + 8 * hi + i];
                ap1[i]     = pp[32 + 8 * hi + i];
                ap1[i + 8] = pp[48 + 8 * hi + i];
            }
        }
        #pragma unroll
        for (int n = 0; n < 4; ++n) {
            v16h bv0, bv1;
            const _Float16* vp = &VldsT[n * 16 + r][0];
            #pragma unroll
            for (int i = 0; i < 16; ++i) {
                bv0[i] = vp[(i < 8 ? 8 * hi + i : 16 + 8 * hi + (i - 8))];
                bv1[i] = vp[32 + (i < 8 ? 8 * hi + i : 16 + 8 * hi + (i - 8))];
            }
            accO[n] = WMMA_F16(ap0, bv0, accO[n]);
            accO[n] = WMMA_F16(ap1, bv1, accO[n]);
        }
    }

    // ---- normalize and write context (merged-heads [B,S,D] layout, f16) ----
    #pragma unroll
    for (int n = 0; n < 4; ++n) {
        #pragma unroll
        for (int v = 0; v < 8; ++v) {
            int qm = v + 8 * hi;
            float o = accO[n][v] / lrow[v];
            size_t row = (size_t)(b * SEQ + qBase + w * 16 + qm);
            Ctx[row * DMODEL + hcol + n * 16 + r] = (_Float16)o;
        }
    }
}

// ---------------------------------------------------------------------------
// Launch: Q/K/V projections -> flash attention -> output projection.
// Inputs: 0=query 1=key 2=value 3=mask 4=Wq 5=bq 6=Wk 7=bk 8=Wv 9=bv 10=Wo 11=bo
// ---------------------------------------------------------------------------
extern "C" void kernel_launch(void* const* d_in, const int* in_sizes, int n_in,
                              void* d_out, int out_size, void* d_ws, size_t ws_size,
                              hipStream_t stream) {
    (void)in_sizes; (void)n_in; (void)out_size; (void)ws_size;

    const float* q  = (const float*)d_in[0];
    const float* k  = (const float*)d_in[1];
    const float* v  = (const float*)d_in[2];
    const unsigned char* mask = (const unsigned char*)d_in[3];
    const float* Wq = (const float*)d_in[4];
    const float* bq = (const float*)d_in[5];
    const float* Wk = (const float*)d_in[6];
    const float* bk = (const float*)d_in[7];
    const float* Wv = (const float*)d_in[8];
    const float* bv = (const float*)d_in[9];
    const float* Wo = (const float*)d_in[10];
    const float* bo = (const float*)d_in[11];
    float* out = (float*)d_out;

    const int M = BATCH * SEQ;                 // 4096 flattened rows
    const size_t elems = (size_t)M * DMODEL;   // 4M f16 each

    _Float16* Qh = (_Float16*)d_ws;
    _Float16* Kh = Qh + elems;
    _Float16* Vh = Kh + elems;
    _Float16* Ch = Vh + elems;                 // 4 * 8MB = 32MB workspace

    dim3 gblock(128);
    dim3 ggrid(DMODEL / 64, M / 64);           // (16, 64)

    gemm_wmma<float, _Float16><<<ggrid, gblock, 0, stream>>>(q, Wq, bq, Qh, M, DMODEL, DMODEL);
    gemm_wmma<float, _Float16><<<ggrid, gblock, 0, stream>>>(k, Wk, bk, Kh, M, DMODEL, DMODEL);
    gemm_wmma<float, _Float16><<<ggrid, gblock, 0, stream>>>(v, Wv, bv, Vh, M, DMODEL, DMODEL);

    dim3 agrid(SEQ / 64, BATCH * HEADS);       // (32, 32)
    attn_wmma<<<agrid, gblock, 0, stream>>>(Qh, Kh, Vh, mask, Ch);

    gemm_wmma<_Float16, float><<<ggrid, gblock, 0, stream>>>(Ch, Wo, bo, out, M, DMODEL, DMODEL);
}
